// RepformerLayer_28492813042214
// MI455X (gfx1250) — matrix operand order repaired and checked by reading, not compiled, not run on hardware
//
#include <hip/hip_runtime.h>

// ---------------------------------------------------------------------------
// Repformer attention layer for MI455X (gfx1250, wave32, WMMA).
// nf=1, nloc=1024, nnei=128, ng2=32, nd=32, nh=4.
// One workgroup (8 wave32s) per loc; all contractions via
// v_wmma_f32_16x16x32_f16 (f32 accumulate). All WMMA fragment reads are
// per-lane-contiguous (2 x b128 per fragment) thanks to N-major operand
// layouts; weights are pre-transposed to f16 in a prep kernel.
// ---------------------------------------------------------------------------

typedef __attribute__((ext_vector_type(16))) _Float16 v16h;
typedef __attribute__((ext_vector_type(8)))  _Float16 v8h;
typedef __attribute__((ext_vector_type(8)))  float    v8f;

#define NLOC 1024
#define NNEI 128
#define NG2  32
#define ND   32
#define NH   4

// ---- LDS layout (bytes), total 160768 ----
// [0,      65536)  Af32 : 128x128 f32 pre-softmax scores (alias: Tf16 128x32 f16)
// [65536, 98304)   U    : per-head q|k f16 (128x64) / final A f16 (128x128)
// [98304, 131072)  g2vT : f16 [h][g][k] = (g2 @ w_v) transposed per head
// [131072,139264)  g2h  : 128x32 f16 staged g2
// [139264,155648)  gacc : 128x32 f32 g2_out accumulator
// [155648,157696)  h2s  : 128x4 f32 (padded h2)
// [157696,158208)  sws  : 128 f32
// [158208,158720)  mks  : 128 f32 (mask as 0/1)
// [158720,160768)  hacc : 128x4 f32 h2_out accumulator
#define SMEM_BYTES 160768

// ---- workspace layout (halves) ----
// wqkT [0,8192)     : [e=0..7][d=0..31][k=0..31]   e = h (q) or 4+h (k)
// wvT  [8192,12288) : [n=0..127][k=0..31]
// whT  [12288,16384): [h=0..3][n=0..31][g=0..31]

// A fragment 16x32 f16: lane lo = row M, half j = K = j + 16*hi
// (matches load_b_nk): one contiguous 32B run per lane.
__device__ __forceinline__ v16h load_a_vec(const _Float16* base, int ld, int lo, int hi) {
  const _Float16* p = base + lo * ld + 16 * hi;
  v8h c0 = *(const v8h*)(p);
  v8h c1 = *(const v8h*)(p + 8);
  return __builtin_shufflevector(c0, c1, 0, 1, 2, 3, 4, 5, 6, 7,
                                 8, 9, 10, 11, 12, 13, 14, 15);
}

// B fragment 32x16 f16 from an N-major (N x K) source: lane lo = column N,
// half j = row K = j + 16*hi: one contiguous 32B run per lane.
__device__ __forceinline__ v16h load_b_nk(const _Float16* base, int ld, int lo, int hi) {
  const _Float16* p = base + lo * ld + 16 * hi;
  v8h c0 = *(const v8h*)(p);
  v8h c1 = *(const v8h*)(p + 8);
  return __builtin_shufflevector(c0, c1, 0, 1, 2, 3, 4, 5, 6, 7,
                                 8, 9, 10, 11, 12, 13, 14, 15);
}

__device__ __forceinline__ v8f wmma32(v16h a, v16h b, v8f c) {
  return __builtin_amdgcn_wmma_f32_16x16x32_f16(false, a, false, b, (short)0, c,
                                                false, false);
}

// ---- prep: convert + transpose shared weights to f16 in workspace ----
__global__ void prep_weights(const float* __restrict__ wqk,
                             const float* __restrict__ wv,
                             const float* __restrict__ wh,
                             _Float16* __restrict__ ws) {
  int tid = blockIdx.x * blockDim.x + threadIdx.x;
  int stride = gridDim.x * blockDim.x;
  // wqkT[(e*32 + d)*32 + k] = wqk[k*256 + d*8 + e]
  for (int i = tid; i < 8192; i += stride) {
    int k = i & 31, d = (i >> 5) & 31, e = i >> 10;
    ws[i] = (_Float16)wqk[k * 256 + d * 8 + e];
  }
  // wvT[n*32 + k] = wv[k*128 + n]
  for (int i = tid; i < 4096; i += stride) {
    int k = i & 31, n = i >> 5;
    ws[8192 + i] = (_Float16)wv[k * 128 + n];
  }
  // whT[(h*32 + n)*32 + g] = wh[(g*4 + h)*32 + n]
  for (int i = tid; i < 4096; i += stride) {
    int g = i & 31, n = (i >> 5) & 31, h = i >> 10;
    ws[12288 + i] = (_Float16)wh[(g * 4 + h) * 32 + n];
  }
}

__global__ void __launch_bounds__(256, 1)
repformer_attn(const float* __restrict__ g2,
               const float* __restrict__ h2,
               const float* __restrict__ sw,
               const int*   __restrict__ nmask,
               const _Float16* __restrict__ wqkT,
               const _Float16* __restrict__ wvT,
               const _Float16* __restrict__ whT,
               const float* __restrict__ b_head,
               const float* __restrict__ w_eqhead,
               float* __restrict__ out) {
  extern __shared__ __align__(16) char smem[];
  float*    Af32 = (float*)(smem);
  _Float16* Tf16 = (_Float16*)(smem);            // alias of Af32 (disjoint lifetime)
  _Float16* U    = (_Float16*)(smem + 65536);
  _Float16* g2vT = (_Float16*)(smem + 98304);
  _Float16* g2h  = (_Float16*)(smem + 131072);
  float*    gacc = (float*)(smem + 139264);
  float*    h2s  = (float*)(smem + 155648);
  float*    sws  = (float*)(smem + 157696);
  float*    mks  = (float*)(smem + 158208);
  float*    hacc = (float*)(smem + 158720);

  const int loc  = blockIdx.x;
  const int tid  = threadIdx.x;
  const int wave = tid >> 5;   // 0..7
  const int lane = tid & 31;
  const int lo   = lane & 15;
  const int hi   = lane >> 4;

  if (tid == 0) {
    __builtin_prefetch(wqkT, 0, 1);
    __builtin_prefetch(wvT, 0, 1);
    __builtin_prefetch(whT, 0, 1);
  }

  // ---- phase 0: stage inputs ----
  const float* g2p = g2 + (size_t)loc * NNEI * NG2;
  for (int i = tid; i < NNEI * NG2; i += 256) g2h[i] = (_Float16)g2p[i];
  const float* h2p = h2 + (size_t)loc * NNEI * 3;
  for (int i = tid; i < NNEI; i += 256) {
    h2s[i * 4 + 0] = h2p[i * 3 + 0];
    h2s[i * 4 + 1] = h2p[i * 3 + 1];
    h2s[i * 4 + 2] = h2p[i * 3 + 2];
    h2s[i * 4 + 3] = 0.f;
    sws[i] = sw[(size_t)loc * NNEI + i];
    mks[i] = nmask[(size_t)loc * NNEI + i] ? 1.f : 0.f;
  }
  for (int i = tid; i < NNEI * NG2; i += 256) gacc[i] = 0.f;
  for (int i = tid; i < NNEI * 4; i += 256) hacc[i] = 0.f;
  __syncthreads();

  // ---- phase 1: g2v = g2 @ w_v -> store transposed: g2vT[(h*32+g)*128 + k] ----
  {
    const int tm = wave;
    v16h a = load_a_vec(g2h + tm * 16 * NG2, NG2, lo, hi);
#pragma unroll
    for (int tn = 0; tn < 8; ++tn) {
      v16h b = load_b_nk(wvT + tn * 16 * 32, 32, lo, hi);
      v8f c = {};
      c = wmma32(a, b, c);
      const int col = tn * 16 + lo;           // col = g*4 + h
      const int gh = (col & 3) * 32 + (col >> 2);
#pragma unroll
      for (int v = 0; v < 8; ++v) {
        int k = tm * 16 + v + 8 * hi;
        g2vT[gh * 128 + k] = (_Float16)c[v];
      }
    }
  }
  __syncthreads();

  const float inv_sqrt_nd = 0.17677669529663687f;   // 1/sqrt(32)
  const float inv_sqrt3   = 0.57735026918962576f;   // 1/sqrt(3)

  for (int h = 0; h < NH; ++h) {
    // ---- (a) q,k projection from transposed wqkT ----
    {
      const int tm = wave;
      v16h a = load_a_vec(g2h + tm * 16 * NG2, NG2, lo, hi);
#pragma unroll
      for (int tn = 0; tn < 2; ++tn) {
        v16h bq = load_b_nk(wqkT + (h * 32 + tn * 16) * 32, 32, lo, hi);
        v8f cq = {};
        cq = wmma32(a, bq, cq);
#pragma unroll
        for (int v = 0; v < 8; ++v)
          U[(tm * 16 + v + 8 * hi) * 64 + tn * 16 + lo] = (_Float16)cq[v];
        v16h bk = load_b_nk(wqkT + ((NH + h) * 32 + tn * 16) * 32, 32, lo, hi);
        v8f ck = {};
        ck = wmma32(a, bk, ck);
#pragma unroll
        for (int v = 0; v < 8; ++v)
          U[(tm * 16 + v + 8 * hi) * 64 + 32 + tn * 16 + lo] = (_Float16)ck[v];
      }
    }
    __syncthreads();

    // ---- (b) S = q @ k^T / sqrt(nd), fused h2ht & sw envelope -> Af32 ----
    {
      const int tm = wave;
      v16h a = load_a_vec(U + tm * 16 * 64, 64, lo, hi);
#pragma unroll
      for (int tn = 0; tn < 8; ++tn) {
        // k stored key-major: key row lo, d contiguous -> N-major load
        v16h b = load_b_nk(U + (tn * 16) * 64 + 32, 64, lo, hi);
        v8f c = {};
        c = wmma32(a, b, c);
#pragma unroll
        for (int v = 0; v < 8; ++v) {
          int q  = tm * 16 + v + 8 * hi;
          int kk = tn * 16 + lo;
          float s  = c[v] * inv_sqrt_nd;
          float hh = h2s[q * 4 + 0] * h2s[kk * 4 + 0] +
                     h2s[q * 4 + 1] * h2s[kk * 4 + 1] +
                     h2s[q * 4 + 2] * h2s[kk * 4 + 2];
          float av = s * hh;
          av = (av + 20.f) * sws[q] * sws[kk] - 20.f;
          Af32[q * 128 + kk] = av;
        }
      }
    }
    __syncthreads();

    // ---- (c) row softmax + mask/sw/h2ht post-scale -> Af16 in U; h2_out acc ----
    if (tid < NNEI) {
      const int q = tid;
      float* row = Af32 + q * 128;
      float m = -3.0e38f;
      for (int kk = 0; kk < 128; ++kk) m = fmaxf(m, row[kk]);
      float ssum = 0.f;
      for (int kk = 0; kk < 128; ++kk) {
        float e = __expf(row[kk] - m);
        ssum += e;
        row[kk] = e;                       // thread-owned row: safe store-back
      }
      float inv = 1.f / ssum;
      float fq = mks[q] * sws[q];
      float hx = 0.f, hy = 0.f, hz = 0.f;
      for (int kk = 0; kk < 128; ++kk) {
        float p  = row[kk] * inv;
        float hh = h2s[q * 4 + 0] * h2s[kk * 4 + 0] +
                   h2s[q * 4 + 1] * h2s[kk * 4 + 1] +
                   h2s[q * 4 + 2] * h2s[kk * 4 + 2];
        p *= fq * mks[kk] * sws[kk] * hh * inv_sqrt3;
        U[q * 128 + kk] = (_Float16)p;
        hx += p * h2s[kk * 4 + 0];
        hy += p * h2s[kk * 4 + 1];
        hz += p * h2s[kk * 4 + 2];
      }
      float we = w_eqhead[h];
      hacc[q * 4 + 0] += we * hx;
      hacc[q * 4 + 1] += we * hy;
      hacc[q * 4 + 2] += we * hz;
    }
    __syncthreads();

    // ---- (d) T = A_h (128x128) @ g2v_h (128x32) -> Tf16 (aliases Af32) ----
    {
      const int tm = wave;
      v8f c0 = {}, c1 = {};
#pragma unroll
      for (int kc = 0; kc < 4; ++kc) {
        v16h a = load_a_vec(U + tm * 16 * 128 + kc * 32, 128, lo, hi);
        v16h b0 = load_b_nk(g2vT + (h * 32) * 128 + kc * 32, 128, lo, hi);
        c0 = wmma32(a, b0, c0);
        v16h b1 = load_b_nk(g2vT + (h * 32 + 16) * 128 + kc * 32, 128, lo, hi);
        c1 = wmma32(a, b1, c1);
      }
#pragma unroll
      for (int v = 0; v < 8; ++v) {
        int q = tm * 16 + v + 8 * hi;
        Tf16[q * 32 + lo]      = (_Float16)c0[v];
        Tf16[q * 32 + 16 + lo] = (_Float16)c1[v];
      }
    }
    __syncthreads();

    // ---- (e) gacc += T @ W_head[g*NH+h, :]  (K = g over 32) ----
    {
      const int tm = wave;
      v16h a = load_a_vec(Tf16 + tm * 16 * 32, 32, lo, hi);
#pragma unroll
      for (int tn = 0; tn < 2; ++tn) {
        v16h b = load_b_nk(whT + (h * 32 + tn * 16) * 32, 32, lo, hi);
        v8f c;
#pragma unroll
        for (int v = 0; v < 8; ++v)
          c[v] = gacc[(tm * 16 + v + 8 * hi) * 32 + tn * 16 + lo];
        c = wmma32(a, b, c);
#pragma unroll
        for (int v = 0; v < 8; ++v)
          gacc[(tm * 16 + v + 8 * hi) * 32 + tn * 16 + lo] = c[v];
      }
    }
    __syncthreads();
  }

  // ---- phase 3: write outputs (g2_out then h2_out, concatenated) ----
  float* og = out + (size_t)loc * NNEI * NG2;
  for (int i = tid; i < NNEI * NG2; i += 256) og[i] = gacc[i] + b_head[i & 31];
  float* oh = out + (size_t)NLOC * NNEI * NG2 + (size_t)loc * NNEI * 3;
  for (int i = tid; i < NNEI * 3; i += 256) {
    int q = i / 3, c = i - q * 3;
    oh[i] = hacc[q * 4 + c];
  }
}

extern "C" void kernel_launch(void* const* d_in, const int* in_sizes, int n_in,
                              void* d_out, int out_size, void* d_ws, size_t ws_size,
                              hipStream_t stream) {
  const float* g2  = (const float*)d_in[0];
  const float* h2  = (const float*)d_in[1];
  const float* sw  = (const float*)d_in[2];
  const float* wqk = (const float*)d_in[3];
  const float* wv  = (const float*)d_in[4];
  const float* wh  = (const float*)d_in[5];
  const float* bh  = (const float*)d_in[6];
  const float* weq = (const float*)d_in[7];
  const int*   nm  = (const int*)d_in[8];

  _Float16* ws16 = (_Float16*)d_ws;

  (void)hipFuncSetAttribute((const void*)repformer_attn,
                            hipFuncAttributeMaxDynamicSharedMemorySize,
                            (int)SMEM_BYTES);

  prep_weights<<<dim3(16), dim3(256), 0, stream>>>(wqk, wv, wh, ws16);
  repformer_attn<<<dim3(NLOC), dim3(256), SMEM_BYTES, stream>>>(
      g2, h2, sw, nm, ws16, ws16 + 8192, ws16 + 12288, bh, weq, (float*)d_out);
}
